// LidarGatLstmNet_51273319579914
// MI455X (gfx1250) — matrix operand order repaired and checked by reading, not compiled, not run on hardware
//
#include <hip/hip_runtime.h>
#include <hip/hip_bf16.h>
#include <math.h>

// ---------------- problem constants (from reference) ----------------
#define T_    6
#define B_    8
#define NPG_  1080
#define N_    (B_ * NPG_)      // 8640 nodes per timestep
#define DEG_  12
#define E_    (N_ * DEG_)      // 103680 raw edges per timestep
#define EPT_  (E_ + N_)        // 112320 edges incl. self-loops
#define NT_   (T_ * N_)        // 51840 flattened nodes
#define ET_   (T_ * EPT_)      // 673920 flattened edges
#define LSTMH 128

typedef __attribute__((ext_vector_type(16))) __bf16 v16bf;
typedef __attribute__((ext_vector_type(8)))  __bf16 v8bf;
typedef __attribute__((ext_vector_type(8)))  float  v8f;

// ---------------- small utility kernels ----------------
__global__ void k_fill(float* __restrict__ p, float v, int n) {
    int i = blockIdx.x * blockDim.x + threadIdx.x;
    if (i < n) p[i] = v;
}

// pack A: f32 [M,K] -> bf16 [M,Kpad], zero padded
__global__ void k_pack_a(const float* __restrict__ A, int K, int Kpad,
                         __bf16* __restrict__ out, int M) {
    int i = blockIdx.x * blockDim.x + threadIdx.x;
    if (i >= M * Kpad) return;
    int m = i / Kpad, k = i - m * Kpad;
    float v = (k < K) ? A[(size_t)m * K + k] : 0.0f;
    out[i] = (__bf16)v;
}

// pack W: f32 [K,F] -> transposed bf16 [Fpad,Kpad], zero padded
__global__ void k_pack_w(const float* __restrict__ W, int K, int F,
                         int Kpad, int Fpad, __bf16* __restrict__ out) {
    int i = blockIdx.x * blockDim.x + threadIdx.x;
    if (i >= Fpad * Kpad) return;
    int n = i / Kpad, k = i - n * Kpad;
    float v = (n < F && k < K) ? W[(size_t)k * F + n] : 0.0f;
    out[i] = (__bf16)v;
}

// ---------------- WMMA GEMM: out[M,F] = Abf[M,Kpad] @ Wt[Fpad,Kpad]^T --------
// 8 waves / block; each wave computes a 16x64 strip (4 N-tiles reuse 1 A frag).
template <int KPAD>
__global__ void k_gemm_wmma(const __bf16* __restrict__ Abf,
                            const __bf16* __restrict__ Wt,
                            float* __restrict__ out, int F) {
    const int wv   = threadIdx.x >> 5;
    const int lane = threadIdx.x & 31;
    const int kh   = lane >> 4;        // half-wave 0/1
    const int l15  = lane & 15;
    const int mt   = blockIdx.x * 8 + wv;
    const int m0   = mt * 16;
    const int n0   = blockIdx.y * 64;
    const int mrow = m0 + l15;

    const __bf16* arow = Abf + (size_t)mrow * KPAD;
    const __bf16* brow0 = Wt + (size_t)(n0 + l15) * KPAD;

    v8f acc0 = {}, acc1 = {}, acc2 = {}, acc3 = {};
#pragma unroll 2
    for (int kb = 0; kb < KPAD; kb += 32) {
        // A frag (ISA 7.12.2 16-bit A 16x32): V0..3 <- kh*8+0..7, V4..7 <- 16+kh*8+0..7
        v8bf a_lo = *(const v8bf*)(arow + kb + kh * 8);
        v8bf a_hi = *(const v8bf*)(arow + kb + 16 + kh * 8);
        v16bf af = __builtin_shufflevector(a_lo, a_hi,
            0, 1, 2, 3, 4, 5, 6, 7, 8, 9, 10, 11, 12, 13, 14, 15);
        // B frags: lane = column, V0..3 <- kh*8+0..7, V4..7 <- 16+kh*8+0..7 (same K order as A)
#pragma unroll
        for (int j = 0; j < 4; ++j) {
            const __bf16* bcol = brow0 + (size_t)j * 16 * KPAD + kb + kh * 8;
            v8bf b_lo = *(const v8bf*)(bcol);
            v8bf b_hi = *(const v8bf*)(bcol + 16);
            v16bf bf = __builtin_shufflevector(b_lo, b_hi,
                0, 1, 2, 3, 4, 5, 6, 7, 8, 9, 10, 11, 12, 13, 14, 15);
            if (j == 0)      acc0 = __builtin_amdgcn_wmma_f32_16x16x32_bf16(false, af, false, bf, (short)0, acc0, false, false);
            else if (j == 1) acc1 = __builtin_amdgcn_wmma_f32_16x16x32_bf16(false, af, false, bf, (short)0, acc1, false, false);
            else if (j == 2) acc2 = __builtin_amdgcn_wmma_f32_16x16x32_bf16(false, af, false, bf, (short)0, acc2, false, false);
            else             acc3 = __builtin_amdgcn_wmma_f32_16x16x32_bf16(false, af, false, bf, (short)0, acc3, false, false);
        }
    }
    // D layout: VGPR r -> row m0 + kh*8 + r, col = n0 + j*16 + l15
#pragma unroll
    for (int j = 0; j < 4; ++j) {
        const v8f& a = (j == 0) ? acc0 : (j == 1) ? acc1 : (j == 2) ? acc2 : acc3;
        int nc = n0 + j * 16 + l15;
        if (nc < F) {
#pragma unroll
            for (int r = 0; r < 8; ++r)
                out[(size_t)(m0 + kh * 8 + r) * F + nc] = a[r];
        }
    }
}

// ---------------- attention scores: es/ed[n,h] = sum_c h[n,h*C+c]*a[h,c] ----
__global__ void k_scores(const float* __restrict__ hfe,
                         const float* __restrict__ as_, const float* __restrict__ ad_,
                         int H, int C, int F,
                         float* __restrict__ es, float* __restrict__ ed) {
    int idx = blockIdx.x * blockDim.x + threadIdx.x;
    if (idx >= NT_ * H) return;
    int n = idx / H, h = idx - n * H;
    const float* row = hfe + (size_t)n * F + h * C;
    float s0 = 0.f, s1 = 0.f;
    for (int c = 0; c < C; ++c) {
        float v = row[c];
        s0 += v * as_[h * C + c];
        s1 += v * ad_[h * C + c];
    }
    es[idx] = s0;
    ed[idx] = s1;
}

// ---------------- edge decode (once, reused by all 4 layers) ----------------
__global__ void k_edge_decode(const int* __restrict__ ei,
                              int* __restrict__ srcA, int* __restrict__ dstA) {
    int g = blockIdx.x * blockDim.x + threadIdx.x;
    if (g >= ET_) return;
    int t = g / EPT_;
    int k = g - t * EPT_;
    int s, d;
    if (k < E_) {
        s = ei[(t * 2 + 0) * E_ + k];
        d = ei[(t * 2 + 1) * E_ + k];
    } else {
        s = d = k - E_;                 // self loop
    }
    srcA[g] = s + t * N_;
    dstA[g] = d + t * N_;
}

__device__ __forceinline__ float lrelu02(float x) {
    return x > 0.f ? x : 0.2f * x;
}

__device__ __forceinline__ void atomicMaxF(float* addr, float val) {
    if (val >= 0.f) atomicMax((int*)addr, __float_as_int(val));
    else            atomicMin((unsigned int*)addr, (unsigned int)__float_as_int(val));
}

// segment max of leaky-relu attention logits
__global__ void k_edge_max(const int* __restrict__ srcA, const int* __restrict__ dstA,
                           const float* __restrict__ es, const float* __restrict__ ed,
                           int H, float* __restrict__ m) {
    int idx = blockIdx.x * blockDim.x + threadIdx.x;
    if (idx >= ET_ * H) return;
    int g = idx / H, h = idx - g * H;
    int src = srcA[g], dst = dstA[g];
    float e = lrelu02(es[src * H + h] + ed[dst * H + h]);
    atomicMaxF(&m[dst * H + h], e);
}

// p = exp(e - m[dst]); s[dst] += p
__global__ void k_edge_exp(const int* __restrict__ srcA, const int* __restrict__ dstA,
                           const float* __restrict__ es, const float* __restrict__ ed,
                           const float* __restrict__ m, int H,
                           float* __restrict__ p, float* __restrict__ s) {
    int idx = blockIdx.x * blockDim.x + threadIdx.x;
    if (idx >= ET_ * H) return;
    int g = idx / H, h = idx - g * H;
    int src = srcA[g], dst = dstA[g];
    float e  = lrelu02(es[src * H + h] + ed[dst * H + h]);
    float pe = __expf(e - m[dst * H + h]);
    p[idx] = pe;
    atomicAdd(&s[dst * H + h], pe);
}

// out[dst, f..f+3] += (p/s[dst]) * hfeat[src, f..f+3]   (float4 per thread)
__global__ void k_edge_msg(const int* __restrict__ srcA, const int* __restrict__ dstA,
                           const float* __restrict__ hfe,
                           const float* __restrict__ p, const float* __restrict__ s,
                           int H, int C, int F, float* __restrict__ out) {
    const int F4 = F >> 2;
    long long idx = (long long)blockIdx.x * blockDim.x + threadIdx.x;
    if (idx >= (long long)ET_ * F4) return;
    int g  = (int)(idx / F4);
    int f  = (int)(idx - (long long)g * F4) << 2;
    int h  = f / C;                            // C is a multiple of 4
    int src = srcA[g], dst = dstA[g];
    float alpha = p[(size_t)g * H + h] / s[dst * H + h];
    const float4 hv = *(const float4*)(hfe + (size_t)src * F + f);
    float* o = out + (size_t)dst * F + f;
    atomicAdd(o + 0, alpha * hv.x);
    atomicAdd(o + 1, alpha * hv.y);
    atomicAdd(o + 2, alpha * hv.z);
    atomicAdd(o + 3, alpha * hv.w);
}

// out = elu(out + bias)
__global__ void k_bias_elu(float* __restrict__ out, const float* __restrict__ b, int F) {
    long long idx = (long long)blockIdx.x * blockDim.x + threadIdx.x;
    if (idx >= (long long)NT_ * F) return;
    int f = (int)(idx % F);
    float v = out[idx] + b[f];
    out[idx] = v > 0.f ? v : expm1f(v);
}

// mean pool per (t, graph): emb[t*B+b, j] += h[n, j] / NPG
__global__ void k_pool(const float* __restrict__ h, float* __restrict__ emb) {
    int idx = blockIdx.x * blockDim.x + threadIdx.x;
    if (idx >= NT_ * 4) return;
    int n = idx >> 2, j = idx & 3;
    int t = n / N_;
    int b = (n - t * N_) / NPG_;
    atomicAdd(&emb[((t * B_) + b) * 4 + j], h[idx] * (1.0f / (float)NPG_));
}

// ---------------- LSTM (T steps, B=8, H=128) + FC head, single block ----------
__device__ __forceinline__ float sigmf(float x) { return 1.0f / (1.0f + __expf(-x)); }

__global__ void k_lstm_fc(const float* __restrict__ emb,
                          const float* __restrict__ Wih, const float* __restrict__ Whh,
                          const float* __restrict__ bih, const float* __restrict__ bhh,
                          const float* __restrict__ Wfc, const float* __restrict__ bfc,
                          float* __restrict__ out) {
    __shared__ float sh[B_ * LSTMH];
    __shared__ float sc[B_ * LSTMH];
    __shared__ float sg[B_ * 4 * LSTMH];
    const int tid = threadIdx.x;

    for (int i = tid; i < B_ * LSTMH; i += blockDim.x) { sh[i] = 0.f; sc[i] = 0.f; }
    __syncthreads();

    for (int t = 0; t < T_; ++t) {
        for (int i = tid; i < B_ * 4 * LSTMH; i += blockDim.x) {
            int b = i / (4 * LSTMH), j = i - b * (4 * LSTMH);
            float acc = bih[j] + bhh[j];
            const float* x = &emb[((t * B_) + b) * 4];
#pragma unroll
            for (int q = 0; q < 4; ++q) acc += x[q] * Wih[j * 4 + q];
            const float* hrow = &sh[b * LSTMH];
            const float* wrow = &Whh[(size_t)j * LSTMH];
            for (int k = 0; k < LSTMH; ++k) acc += hrow[k] * wrow[k];
            sg[i] = acc;
        }
        __syncthreads();
        for (int i = tid; i < B_ * LSTMH; i += blockDim.x) {
            int b = i / LSTMH, k = i - b * LSTMH;
            const float* g = &sg[b * 4 * LSTMH];
            float iv = sigmf(g[k]);
            float fv = sigmf(g[LSTMH + k]);
            float gv = tanhf(g[2 * LSTMH + k]);
            float ov = sigmf(g[3 * LSTMH + k]);
            float c  = fv * sc[i] + iv * gv;
            sc[i] = c;
            sh[i] = ov * tanhf(c);
        }
        __syncthreads();
    }
    for (int i = tid; i < B_ * 2; i += blockDim.x) {
        int b = i >> 1, o = i & 1;
        float acc = bfc[o];
        const float* hrow = &sh[b * LSTMH];
        const float* wrow = &Wfc[o * LSTMH];
        for (int k = 0; k < LSTMH; ++k) acc += hrow[k] * wrow[k];
        out[b * 2 + o] = acc;
    }
}

// ---------------- host-side orchestration ----------------
static inline int cdiv(long long a, int b) { return (int)((a + b - 1) / b); }

extern "C" void kernel_launch(void* const* d_in, const int* in_sizes, int n_in,
                              void* d_out, int out_size, void* d_ws, size_t ws_size,
                              hipStream_t stream) {
    (void)in_sizes; (void)n_in; (void)out_size; (void)ws_size;
    const float* x_seq = (const float*)d_in[0];
    const int*   ei    = (const int*)d_in[1];

    const float* Wl[4]  = {(const float*)d_in[3],  (const float*)d_in[7],
                           (const float*)d_in[11], (const float*)d_in[15]};
    const float* asl[4] = {(const float*)d_in[4],  (const float*)d_in[8],
                           (const float*)d_in[12], (const float*)d_in[16]};
    const float* adl[4] = {(const float*)d_in[5],  (const float*)d_in[9],
                           (const float*)d_in[13], (const float*)d_in[17]};
    const float* bl[4]  = {(const float*)d_in[6],  (const float*)d_in[10],
                           (const float*)d_in[14], (const float*)d_in[18]};
    const float* Wih = (const float*)d_in[19];
    const float* Whh = (const float*)d_in[20];
    const float* bih = (const float*)d_in[21];
    const float* bhh = (const float*)d_in[22];
    const float* Wfc = (const float*)d_in[23];
    const float* bfc = (const float*)d_in[24];

    const int Fin[4]  = {3, 256, 128, 64};
    const int Kpad[4] = {32, 256, 128, 64};
    const int F[4]    = {256, 128, 64, 4};
    const int Fpad[4] = {256, 128, 64, 64};
    const int H[4]    = {8, 8, 8, 1};
    const int C[4]    = {32, 16, 8, 4};

    // workspace layout (float units)
    float*  ws    = (float*)d_ws;
    float*  bufIO = ws;                                  // NT_*256
    float*  bufFe = bufIO + (size_t)NT_ * 256;           // NT_*256
    __bf16* Abf   = (__bf16*)(bufFe + (size_t)NT_ * 256);// NT_*256 bf16
    __bf16* Wtbf  = Abf + (size_t)NT_ * 256;             // 256*256 bf16
    float*  es    = (float*)(Wtbf + 256 * 256);          // NT_*8
    float*  ed    = es + (size_t)NT_ * 8;                // NT_*8
    float*  mbuf  = ed + (size_t)NT_ * 8;                // NT_*8
    float*  sbuf  = mbuf + (size_t)NT_ * 8;              // NT_*8
    float*  pbuf  = sbuf + (size_t)NT_ * 8;              // ET_*8
    float*  emb   = pbuf + (size_t)ET_ * 8;              // T_*B_*4
    int*    srcA  = (int*)(emb + T_ * B_ * 4);           // ET_
    int*    dstA  = srcA + ET_;                          // ET_

    const int TPB = 256;

    // decode edges (incl. self loops) once — reused by all 4 layers
    k_edge_decode<<<cdiv(ET_, TPB), TPB, 0, stream>>>(ei, srcA, dstA);

    const float* in = x_seq;
    for (int l = 0; l < 4; ++l) {
        const int f = F[l], h = H[l], c = C[l], k = Fin[l], kp = Kpad[l], fp = Fpad[l];
        // pack operands to padded bf16
        k_pack_a<<<cdiv((long long)NT_ * kp, TPB), TPB, 0, stream>>>(in, k, kp, Abf, NT_);
        k_pack_w<<<cdiv((long long)fp * kp, TPB), TPB, 0, stream>>>(Wl[l], k, f, kp, fp, Wtbf);
        // h = in @ W  (bf16 WMMA, f32 accumulate); 16x64 strip per wave
        dim3 ggrid(NT_ / 16 / 8, fp / 64);
        if      (kp == 32)  k_gemm_wmma<32> <<<ggrid, 256, 0, stream>>>(Abf, Wtbf, bufFe, f);
        else if (kp == 64)  k_gemm_wmma<64> <<<ggrid, 256, 0, stream>>>(Abf, Wtbf, bufFe, f);
        else if (kp == 128) k_gemm_wmma<128><<<ggrid, 256, 0, stream>>>(Abf, Wtbf, bufFe, f);
        else                k_gemm_wmma<256><<<ggrid, 256, 0, stream>>>(Abf, Wtbf, bufFe, f);
        // reset segment buffers (bufIO's old contents are dead after the pack)
        k_fill<<<cdiv(NT_ * h, TPB), TPB, 0, stream>>>(mbuf, -3.0e38f, NT_ * h);
        k_fill<<<cdiv(NT_ * h, TPB), TPB, 0, stream>>>(sbuf, 0.0f, NT_ * h);
        k_fill<<<cdiv((long long)NT_ * f, TPB), TPB, 0, stream>>>(bufIO, 0.0f, NT_ * f);
        // attention softmax over incoming edges
        k_scores<<<cdiv(NT_ * h, TPB), TPB, 0, stream>>>(bufFe, asl[l], adl[l], h, c, f, es, ed);
        k_edge_max<<<cdiv((long long)ET_ * h, TPB), TPB, 0, stream>>>(srcA, dstA, es, ed, h, mbuf);
        k_edge_exp<<<cdiv((long long)ET_ * h, TPB), TPB, 0, stream>>>(srcA, dstA, es, ed, mbuf, h, pbuf, sbuf);
        k_edge_msg<<<cdiv((long long)ET_ * (f >> 2), TPB), TPB, 0, stream>>>(srcA, dstA, bufFe, pbuf, sbuf, h, c, f, bufIO);
        k_bias_elu<<<cdiv((long long)NT_ * f, TPB), TPB, 0, stream>>>(bufIO, bl[l], f);
        in = bufIO;
    }

    // mean pool -> emb[T,B,4]
    k_fill<<<1, TPB, 0, stream>>>(emb, 0.0f, T_ * B_ * 4);
    k_pool<<<cdiv(NT_ * 4, TPB), TPB, 0, stream>>>(bufIO, emb);

    // LSTM over T + FC head -> d_out [B,2]
    k_lstm_fc<<<1, 256, 0, stream>>>(emb, Wih, Whh, bih, bhh, Wfc, bfc, (float*)d_out);
}
